// MultiHeadSelfAttentionBlock_37460704756207
// MI455X (gfx1250) — matrix operand order repaired and checked by reading, not compiled, not run on hardware
//
#include <hip/hip_runtime.h>
#include <hip/hip_bf16.h>

// MI455X (gfx1250) multi-head self-attention, WMMA f16 (fp32 accum).
// B=4, T=2048, E=1024, H=16, D=64.  ~137 GFLOP vs ~100MB traffic => compute
// bound => all GEMMs on v_wmma_f32_16x16x32_f16.  Attention is flash-style
// (online softmax); K/V tiles are staged into LDS with CDNA5 async loads
// (global_load_async_to_lds_b128 + s_wait_asynccnt), double-buffered so the
// DMA for block j+1 overlaps WMMA+softmax of block j, shared by 4 waves.

typedef __attribute__((ext_vector_type(8)))  _Float16 v8h;
typedef __attribute__((ext_vector_type(16))) _Float16 v16h;
typedef __attribute__((ext_vector_type(8)))  float    v8f;

#define TOK   8192      // B*T
#define EMB   1024
#define TSEQ  2048
#define NH    16
#define DH    64

__device__ __forceinline__ v16h ld_pair(const _Float16* p0, const _Float16* p1) {
  union { v16h w; v8h h[2]; } u;
  u.h[0] = *(const v8h*)p0;
  u.h[1] = *(const v8h*)p1;
  return u.w;
}

// A operand (16x32): lane L holds given row, K chunks {c..c+7},{16+c..16+c+7},
// c = 8*(L>>4).  Row-major f16 source, leading dim ldm.
__device__ __forceinline__ v16h loadA(const _Float16* __restrict__ base, int row,
                                      int ldm, int k0, int lane) {
  const _Float16* p = base + (size_t)row * ldm + k0 + 8 * (lane >> 4);
  return ld_pair(p, p + 16);
}

// B operand (32x16): lane L holds column nrow, K chunks {c..c+7},{16+c..16+c+7}, c = 8*(L>>4).
// Source stores each N-column as a contiguous row (leading dim ldm).
__device__ __forceinline__ v16h loadB(const _Float16* __restrict__ base, int nrow,
                                      int ldm, int k0, int lane) {
  const _Float16* p = base + (size_t)nrow * ldm + k0 + 8 * (lane >> 4);
  return ld_pair(p, p + 16);
}

__device__ __forceinline__ v8f wmma32(v16h a, v16h b, v8f c) {
  return __builtin_amdgcn_wmma_f32_16x16x32_f16(false, a, false, b, (short)0, c,
                                                false, false);
}

// CDNA5 async copy: 16B global -> LDS per lane, tracked by ASYNCcnt.
__device__ __forceinline__ void async_b128(unsigned lds_off, const void* gptr) {
  asm volatile("global_load_async_to_lds_b128 %0, %1, off"
               :: "v"(lds_off), "v"((unsigned long long)(size_t)gptr)
               : "memory");
}
#define WAIT_ASYNC() asm volatile("s_wait_asynccnt 0" ::: "memory")
#define WAIT_DS()    asm volatile("s_wait_dscnt 0" ::: "memory")

// Half-wave row reductions (C/D layout: one row = 16 lanes of one half-wave).
__device__ __forceinline__ float redmax16(float v) {
  v = fmaxf(v, __shfl_xor(v, 1, 32));
  v = fmaxf(v, __shfl_xor(v, 2, 32));
  v = fmaxf(v, __shfl_xor(v, 4, 32));
  v = fmaxf(v, __shfl_xor(v, 8, 32));
  return v;
}
__device__ __forceinline__ float redsum16(float v) {
  v += __shfl_xor(v, 1, 32);
  v += __shfl_xor(v, 2, 32);
  v += __shfl_xor(v, 4, 32);
  v += __shfl_xor(v, 8, 32);
  return v;
}

// ---------------- stage 0: fp32 -> f16 ----------------
__global__ void k_cvt(const float* __restrict__ src, _Float16* __restrict__ dst, int n) {
  int i = blockIdx.x * blockDim.x + threadIdx.x;
  int stride = gridDim.x * blockDim.x;
  for (; i < n; i += stride) dst[i] = (_Float16)src[i];
}

// ---------------- stage 1: QKV projection ----------------
// Wave computes a 32x64 macro-tile (2 M-tiles x 4 N-tiles), register
// double-buffered over K so WMMAs overlap the next K-step's loads.
__global__ void __launch_bounds__(128) k_qkv(const _Float16* __restrict__ xh,
                                             const _Float16* __restrict__ wh,
                                             _Float16* __restrict__ Qh,
                                             _Float16* __restrict__ Kh,
                                             _Float16* __restrict__ Vth) {
  const int lane = threadIdx.x & 31;
  const int wave = blockIdx.x * 4 + (threadIdx.x >> 5);
  const int ng = wave % 48;           // 48 groups of 64 features (3072)
  const int mg = wave / 48;           // 256 groups of 32 tokens
  const int m0 = mg * 32, n0 = ng * 64;
  const int lh = lane >> 4;
  v8f acc[2][4] = {};
  v16h a[2], bm[4];
  a[0] = loadA(xh, m0 + (lane & 15), EMB, 0, lane);
  a[1] = loadA(xh, m0 + 16 + (lane & 15), EMB, 0, lane);
#pragma unroll
  for (int c = 0; c < 4; c++) bm[c] = loadB(wh, n0 + c * 16 + (lane & 15), EMB, 0, lane);
  for (int k0 = 0; k0 < EMB; k0 += 32) {
    const int kn = (k0 + 32) & (EMB - 1);           // wrap: last preload harmless
    v16h na0 = loadA(xh, m0 + (lane & 15), EMB, kn, lane);
    v16h na1 = loadA(xh, m0 + 16 + (lane & 15), EMB, kn, lane);
    v16h nb[4];
#pragma unroll
    for (int c = 0; c < 4; c++) nb[c] = loadB(wh, n0 + c * 16 + (lane & 15), EMB, kn, lane);
#pragma unroll
    for (int c = 0; c < 4; c++) acc[0][c] = wmma32(a[0], bm[c], acc[0][c]);
#pragma unroll
    for (int c = 0; c < 4; c++) acc[1][c] = wmma32(a[1], bm[c], acc[1][c]);
    a[0] = na0; a[1] = na1;
#pragma unroll
    for (int c = 0; c < 4; c++) bm[c] = nb[c];
  }
#pragma unroll
  for (int mi = 0; mi < 2; mi++) {
#pragma unroll
    for (int c = 0; c < 4; c++) {
      int f = n0 + c * 16 + (lane & 15);
      int which = f >> 10;
      int h = (f >> 6) & 15;
      int d = f & 63;
#pragma unroll
      for (int r = 0; r < 8; r++) {
        int bt = m0 + mi * 16 + r + 8 * lh;
        int b = bt >> 11, t = bt & 2047;
        int bh = b * NH + h;
        _Float16 v = (_Float16)acc[mi][c][r];
        if (which == 0)      Qh[((size_t)bh * TSEQ + t) * DH + d] = v;
        else if (which == 1) Kh[((size_t)bh * TSEQ + t) * DH + d] = v;
        else                 Vth[((size_t)bh * DH + d) * TSEQ + t] = v;
      }
    }
  }
}

// ---------------- stage 2: flash attention ----------------
// Block = one (b,h), 4 waves = 4 adjacent 16-row Q tiles.  K/V j-blocks (32
// wide) staged to LDS via async DMA, double-buffered; all 4 waves share them.
__global__ void __launch_bounds__(128) k_attn(const _Float16* __restrict__ Qh,
                                              const _Float16* __restrict__ Kh,
                                              const _Float16* __restrict__ Vth,
                                              _Float16* __restrict__ Ah) {
  __shared__ alignas(16) _Float16 sK[2][32 * DH];   // 8 KB (rows j, 64 halves)
  __shared__ alignas(16) _Float16 sV[2][DH * 32];   // 8 KB (rows d, 32 halves)
  __shared__ alignas(16) _Float16 sP[4][16 * 32];   // 4 KB per-wave P scratch
  const int tid = threadIdx.x;
  const int lane = tid & 31;
  const int wid = tid >> 5;
  const int bh = blockIdx.x >> 5;          // 64 (b,h)
  const int mtg = blockIdx.x & 31;         // 32 groups of 4 row-tiles
  const int b = bh >> 4, h = bh & 15;
  const _Float16* Qb = Qh + (size_t)bh * TSEQ * DH;
  const _Float16* Kb = Kh + (size_t)bh * TSEQ * DH;
  const _Float16* Vb = Vth + (size_t)bh * DH * TSEQ;
  const int m0 = (mtg * 4 + wid) * 16;
  const int lh = lane >> 4;
  const float scale = 0.125f;              // DH^-0.5

  // cooperative async stage of one 32-wide K/V j-block into LDS buffer `buf`
  auto stage = [&](int jn, int buf) {
    // K tile: rows jn..jn+31 are contiguous (32*64 halves = 4KB flat copy)
    {
      const _Float16* src = Kb + (size_t)jn * DH + tid * 16;
      unsigned dst = (unsigned)(size_t)&sK[buf][tid * 16];
      async_b128(dst, src);
      async_b128(dst + 16, src + 8);
    }
    // V tile: 64 d-rows x 32 halves (V^T rows contiguous in t)
    {
      int d = tid >> 1, part = tid & 1;
      const _Float16* src = Vb + (size_t)d * TSEQ + jn + part * 16;
      unsigned dst = (unsigned)(size_t)&sV[buf][d * 32 + part * 16];
      async_b128(dst, src);
      async_b128(dst + 16, src + 8);
    }
  };

  v16h qa0 = loadA(Qb, m0 + (lane & 15), DH, 0, lane);
  v16h qa1 = loadA(Qb, m0 + (lane & 15), DH, 32, lane);

  v8f acc[4] = {};
  float mrow[8], lrow[8];
#pragma unroll
  for (int r = 0; r < 8; r++) { mrow[r] = -3.0e38f; lrow[r] = 0.0f; }
  _Float16* myl = sP[wid];

  stage(0, 0);                             // prologue DMA
  for (int j0 = 0; j0 < TSEQ; j0 += 32) {
    const int cur = (j0 >> 5) & 1;
    WAIT_ASYNC();                          // this buffer's DMA landed
    __syncthreads();                       // everyone done reading buf cur^1
    if (j0 + 32 < TSEQ) stage(j0 + 32, cur ^ 1);   // overlap next DMA

    // S = Q K^T (two 16-col tiles, d reduced in two WMMAs each) from LDS
    v16h b00 = loadB(sK[cur], (lane & 15), DH, 0, lane);
    v16h b01 = loadB(sK[cur], (lane & 15), DH, 32, lane);
    v16h b10 = loadB(sK[cur], 16 + (lane & 15), DH, 0, lane);
    v16h b11 = loadB(sK[cur], 16 + (lane & 15), DH, 32, lane);
    v8f s0 = {}, s1 = {};
    s0 = wmma32(qa0, b00, s0);
    s0 = wmma32(qa1, b01, s0);
    s1 = wmma32(qa0, b10, s1);
    s1 = wmma32(qa1, b11, s1);

    // hoist V operands: ds latency hides under the softmax VALU below
    v16h vb[4];
#pragma unroll
    for (int c = 0; c < 4; c++) vb[c] = loadB(sV[cur], c * 16 + (lane & 15), 32, 0, lane);

    // online softmax per row (rows live in half-waves)
#pragma unroll
    for (int r = 0; r < 8; r++) {
      float a0 = s0[r] * scale, a1 = s1[r] * scale;
      float tmax = redmax16(fmaxf(a0, a1));
      float nm = fmaxf(mrow[r], tmax);
      float alpha = __expf(mrow[r] - nm);
      mrow[r] = nm;
      float p0 = __expf(a0 - nm), p1 = __expf(a1 - nm);
      lrow[r] = lrow[r] * alpha + redsum16(p0 + p1);
#pragma unroll
      for (int c = 0; c < 4; c++) acc[c][r] = acc[c][r] * alpha;
      int ra = (r + 8 * lh) * 32 + (lane & 15);      // row-major [row][j]
      myl[ra] = (_Float16)p0;
      myl[ra + 16] = (_Float16)p1;
    }
    WAIT_DS();                              // P visible for re-layout

    const _Float16* pp = myl + (lane & 15) * 32 + 8 * lh;
    v16h pa = ld_pair(pp, pp + 16);         // P in A-layout (16x32, K=j)

#pragma unroll
    for (int c = 0; c < 4; c++) acc[c] = wmma32(pa, vb[c], acc[c]);
  }

#pragma unroll
  for (int r = 0; r < 8; r++) lrow[r] = 1.0f / lrow[r];
#pragma unroll
  for (int c = 0; c < 4; c++) {
    int f = h * DH + c * 16 + (lane & 15);
#pragma unroll
    for (int r = 0; r < 8; r++) {
      int t = m0 + r + 8 * lh;
      Ah[((size_t)(b * TSEQ + t)) * EMB + f] = (_Float16)(acc[c][r] * lrow[r]);
    }
  }
}

// ---------------- stage 3: output projection ----------------
__global__ void __launch_bounds__(128) k_out(const _Float16* __restrict__ Ah,
                                             const _Float16* __restrict__ Wo,
                                             float* __restrict__ out) {
  const int lane = threadIdx.x & 31;
  const int wave = blockIdx.x * 4 + (threadIdx.x >> 5);
  const int ng = wave & 15;           // 16 groups of 64 outputs
  const int mg = wave >> 4;           // 256 groups of 32 tokens
  const int m0 = mg * 32, n0 = ng * 64;
  const int lh = lane >> 4;
  v8f acc[2][4] = {};
  v16h a[2], bm[4];
  a[0] = loadA(Ah, m0 + (lane & 15), EMB, 0, lane);
  a[1] = loadA(Ah, m0 + 16 + (lane & 15), EMB, 0, lane);
#pragma unroll
  for (int c = 0; c < 4; c++) bm[c] = loadB(Wo, n0 + c * 16 + (lane & 15), EMB, 0, lane);
  for (int k0 = 0; k0 < EMB; k0 += 32) {
    const int kn = (k0 + 32) & (EMB - 1);
    v16h na0 = loadA(Ah, m0 + (lane & 15), EMB, kn, lane);
    v16h na1 = loadA(Ah, m0 + 16 + (lane & 15), EMB, kn, lane);
    v16h nb[4];
#pragma unroll
    for (int c = 0; c < 4; c++) nb[c] = loadB(Wo, n0 + c * 16 + (lane & 15), EMB, kn, lane);
#pragma unroll
    for (int c = 0; c < 4; c++) acc[0][c] = wmma32(a[0], bm[c], acc[0][c]);
#pragma unroll
    for (int c = 0; c < 4; c++) acc[1][c] = wmma32(a[1], bm[c], acc[1][c]);
    a[0] = na0; a[1] = na1;
#pragma unroll
    for (int c = 0; c < 4; c++) bm[c] = nb[c];
  }
#pragma unroll
  for (int mi = 0; mi < 2; mi++)
#pragma unroll
    for (int c = 0; c < 4; c++)
#pragma unroll
      for (int r = 0; r < 8; r++) {
        int bt = m0 + mi * 16 + r + 8 * lh;
        out[(size_t)bt * EMB + n0 + c * 16 + (lane & 15)] = acc[mi][c][r];
      }
}

extern "C" void kernel_launch(void* const* d_in, const int* in_sizes, int n_in,
                              void* d_out, int out_size, void* d_ws, size_t ws_size,
                              hipStream_t stream) {
  const float* x = (const float*)d_in[0];        // (4,2048,1024)
  const float* wqkv = (const float*)d_in[1];     // (3072,1024)
  const float* wout = (const float*)d_in[2];     // (1024,1024)
  float* out = (float*)d_out;

  _Float16* ws = (_Float16*)d_ws;
  const size_t NX = (size_t)TOK * EMB;
  const size_t NWQ = (size_t)3 * EMB * EMB;
  const size_t NWO = (size_t)EMB * EMB;
  const size_t NHD = (size_t)4 * NH * TSEQ * DH;
  _Float16* xh = ws;
  _Float16* wqh = xh + NX;
  _Float16* woh = wqh + NWQ;
  _Float16* Qh = woh + NWO;
  _Float16* Kh = Qh + NHD;
  _Float16* Vth = Kh + NHD;
  _Float16* Ah = Vth + NHD;                      // ~90 MB of ws total

  k_cvt<<<2048, 256, 0, stream>>>(x, xh, (int)NX);
  k_cvt<<<1024, 256, 0, stream>>>(wqkv, wqh, (int)NWQ);
  k_cvt<<<512, 256, 0, stream>>>(wout, woh, (int)NWO);
  k_qkv<<<3072, 128, 0, stream>>>(xh, wqh, Qh, Kh, Vth);   // 256*48 waves
  k_attn<<<2048, 128, 0, stream>>>(Qh, Kh, Vth, Ah);       // 64*32 blocks
  k_out<<<1024, 128, 0, stream>>>(Ah, woh, out);           // 256*16 waves
}
